// Block_70463233458307
// MI455X (gfx1250) — matrix operand order repaired and checked by reading, not compiled, not run on hardware
//
#include <hip/hip_runtime.h>
#include <math.h>

// ---------------------------------------------------------------------------
// RWKV block forward for MI455X (gfx1250, wave32, WMMA bf16 16x16x32).
// Pipeline: LN1 -> mix -> {k,v,r} GEMMs -> WKV scan -> out GEMM (+residual)
//           LN2 -> mix -> fr GEMM(sigmoid) -> fk GEMM(relu^2->bf16)
//           -> value GEMM (out = x1 + fr * vk)
// All GEMMs run through one WMMA-tiled kernel with fused epilogues.
// ---------------------------------------------------------------------------

typedef __attribute__((ext_vector_type(16))) __bf16 v16bf;
typedef __attribute__((ext_vector_type(8)))  __bf16 bf16x8;
typedef __attribute__((ext_vector_type(8)))  float  v8f;

#define LDS_STRIDE 40   // 32 payload + 8 pad halves -> 80B rows, conflict-free 16B ds ops

enum { EP_STORE = 0, EP_SIGMOID = 1, EP_RESADD = 2, EP_RELU2 = 3, EP_FMAOUT = 4 };

// ---------------------- WMMA fragment loaders (LDS) ------------------------
// A (16x32 bf16): lanes 0-15 row m hold K{0..15}; lanes 16-31 K{16..31}
__device__ __forceinline__ v16bf frag_a(const __bf16* As, int m, int half) {
  const __bf16* p = As + m * LDS_STRIDE;
  bf16x8 lo = *(const bf16x8*)(p + half * 16);
  bf16x8 hi = *(const bf16x8*)(p + half * 16 + 8);
  v16bf f;
#pragma unroll
  for (int i = 0; i < 8; ++i) { f[i] = lo[i]; f[i + 8] = hi[i]; }
  return f;
}
// B (32x16 bf16), staged as B^T rows: lane group selects K half, lane&15 selects column n
__device__ __forceinline__ v16bf frag_b(const __bf16* Bs, int n, int kr) {
  const __bf16* p = Bs + n * LDS_STRIDE + kr;
  bf16x8 lo = *(const bf16x8*)(p);
  bf16x8 hi = *(const bf16x8*)(p + 8);
  v16bf f;
#pragma unroll
  for (int i = 0; i < 8; ++i) { f[i] = lo[i]; f[i + 8] = hi[i]; }
  return f;
}

// ---------------------------- tiled WMMA GEMM ------------------------------
// C[M,N] = A[M,K] (bf16, row-major) x Bt[N,K]^T (bf16, pre-transposed weights)
// block: 128x64 tile, 256 threads = 8 waves, each wave 32x32 (2x2 WMMA tiles)
template <int EP>
__global__ __launch_bounds__(256) void gemm_bf16_wmma(
    const __bf16* __restrict__ A, const __bf16* __restrict__ Bt,
    void* __restrict__ Cout, const float* __restrict__ Xin,
    const float* __restrict__ Aux, int N, int K)
{
  __shared__ __align__(16) __bf16 As[128 * LDS_STRIDE];
  __shared__ __align__(16) __bf16 Bs[64 * LDS_STRIDE];

  const int tid   = threadIdx.x;
  const int lane  = tid & 31;
  const int wid   = tid >> 5;
  const int wm    = wid & 3;     // 4 waves along M (4*32 = 128)
  const int wn    = wid >> 2;    // 2 waves along N (2*32 = 64)
  const int bM    = blockIdx.y * 128;
  const int bN    = blockIdx.x * 64;
  const int lrow  = lane & 15;
  const int lhalf = lane >> 4;

  v8f acc[2][2] = {};

  const int arow = tid >> 1, ahalf = tid & 1;

  for (int k0 = 0; k0 < K; k0 += 32) {
    // ---- stage A tile (128x32) ----
    {
      const __bf16* g = A + (size_t)(bM + arow) * K + k0 + ahalf * 16;
      if (k0 + 32 < K) __builtin_prefetch(g + 32, 0, 3);
      bf16x8 v0 = *(const bf16x8*)g;
      bf16x8 v1 = *(const bf16x8*)(g + 8);
      __bf16* s = As + arow * LDS_STRIDE + ahalf * 16;
      *(bf16x8*)s       = v0;
      *(bf16x8*)(s + 8) = v1;
    }
    // ---- stage B^T tile (64x32) ----
    if (tid < 128) {
      const __bf16* g = Bt + (size_t)(bN + arow) * K + k0 + ahalf * 16;
      if (k0 + 32 < K) __builtin_prefetch(g + 32, 0, 3);
      bf16x8 v0 = *(const bf16x8*)g;
      bf16x8 v1 = *(const bf16x8*)(g + 8);
      __bf16* s = Bs + arow * LDS_STRIDE + ahalf * 16;
      *(bf16x8*)s       = v0;
      *(bf16x8*)(s + 8) = v1;
    }
    __syncthreads();

    v16bf a0 = frag_a(As, wm * 32 + lrow,      lhalf);
    v16bf a1 = frag_a(As, wm * 32 + 16 + lrow, lhalf);
    v16bf b0 = frag_b(Bs, wn * 32 + lrow,      lhalf * 16);
    v16bf b1 = frag_b(Bs, wn * 32 + 16 + lrow, lhalf * 16);

    acc[0][0] = __builtin_amdgcn_wmma_f32_16x16x32_bf16(false, a0, false, b0, (short)0, acc[0][0], false, false);
    acc[0][1] = __builtin_amdgcn_wmma_f32_16x16x32_bf16(false, a0, false, b1, (short)0, acc[0][1], false, false);
    acc[1][0] = __builtin_amdgcn_wmma_f32_16x16x32_bf16(false, a1, false, b0, (short)0, acc[1][0], false, false);
    acc[1][1] = __builtin_amdgcn_wmma_f32_16x16x32_bf16(false, a1, false, b1, (short)0, acc[1][1], false, false);
    __syncthreads();
  }

  // ---- epilogue: C layout = 8 VGPRs, lanes 0-15 M base 0, lanes 16-31 M base 8
#pragma unroll
  for (int i = 0; i < 2; ++i) {
#pragma unroll
    for (int j = 0; j < 2; ++j) {
      const int r0 = bM + wm * 32 + i * 16 + lhalf * 8;
      const int c  = bN + wn * 32 + j * 16 + lrow;
#pragma unroll
      for (int e = 0; e < 8; ++e) {
        const size_t idx = (size_t)(r0 + e) * N + c;
        const float v = acc[i][j][e];
        if (EP == EP_STORE) {
          ((float*)Cout)[idx] = v;
        } else if (EP == EP_SIGMOID) {
          ((float*)Cout)[idx] = 1.0f / (1.0f + expf(-v));
        } else if (EP == EP_RESADD) {
          ((float*)Cout)[idx] = Xin[idx] + v;
        } else if (EP == EP_RELU2) {
          const float t = v > 0.0f ? v : 0.0f;
          ((__bf16*)Cout)[idx] = (__bf16)(t * t);
        } else { // EP_FMAOUT
          ((float*)Cout)[idx] = Xin[idx] + Aux[idx] * v;
        }
      }
    }
  }
}

// ------------------------------ LayerNorm ----------------------------------
__global__ __launch_bounds__(256) void ln_kernel(
    const float* __restrict__ x, const float* __restrict__ w,
    const float* __restrict__ b, float* __restrict__ o, int Dd)
{
  __shared__ float red[256];
  const int t = blockIdx.x, tid = threadIdx.x;
  const float* xi = x + (size_t)t * Dd;

  float s = 0.0f;
  for (int d = tid; d < Dd; d += 256) s += xi[d];
  red[tid] = s; __syncthreads();
  for (int st = 128; st > 0; st >>= 1) { if (tid < st) red[tid] += red[tid + st]; __syncthreads(); }
  const float mean = red[0] / (float)Dd;
  __syncthreads();

  float s2 = 0.0f;
  for (int d = tid; d < Dd; d += 256) { const float v = xi[d] - mean; s2 += v * v; }
  red[tid] = s2; __syncthreads();
  for (int st = 128; st > 0; st >>= 1) { if (tid < st) red[tid] += red[tid + st]; __syncthreads(); }
  const float rstd = rsqrtf(red[0] / (float)Dd + 1e-5f);

  float* oi = o + (size_t)t * Dd;
  for (int d = tid; d < Dd; d += 256)
    oi[d] = (xi[d] - mean) * rstd * w[d] + b[d];
}

// -------------------- time-shift mix (f32 -> bf16 operand) -----------------
__global__ __launch_bounds__(256) void mix_kernel(
    const float* __restrict__ xn, const float* __restrict__ mk,
    __bf16* __restrict__ out, int Tlen, int Dd, size_t total)
{
  const size_t stride = (size_t)gridDim.x * blockDim.x;
  for (size_t i = (size_t)blockIdx.x * blockDim.x + threadIdx.x; i < total; i += stride) {
    const int d = (int)(i % Dd);
    const size_t tt = i / Dd;
    const int t = (int)(tt % Tlen);
    const float cur  = xn[i];
    const float prev = (t == 0) ? 0.0f : xn[i - Dd];
    const float m = mk[d];
    out[i] = (__bf16)(cur * m + prev * (1.0f - m));
  }
}

// ------------ weight prep: f32 [K,N] -> bf16 transposed [N,K] --------------
__global__ __launch_bounds__(256) void convT_kernel(
    const float* __restrict__ src, __bf16* __restrict__ dst, int Kd, int Nd)
{
  const size_t total = (size_t)Kd * Nd;
  const size_t stride = (size_t)gridDim.x * blockDim.x;
  for (size_t i = (size_t)blockIdx.x * blockDim.x + threadIdx.x; i < total; i += stride) {
    const size_t k = i / Nd, n = i % Nd;
    dst[n * (size_t)Kd + k] = (__bf16)src[i];
  }
}

// -------------------- WKV recurrence (stable log-space) --------------------
__global__ __launch_bounds__(256) void wkv_kernel(
    const float* __restrict__ Kb, const float* __restrict__ Vb,
    const float* __restrict__ SR, const float* __restrict__ td,
    const float* __restrict__ tf, __bf16* __restrict__ outs,
    int Bn, int Tlen, int Dd)
{
  const int idx = blockIdx.x * blockDim.x + threadIdx.x;
  if (idx >= Bn * Dd) return;
  const int b = idx / Dd, d = idx % Dd;
  const float w = expf(td[d]);
  const float u = tf[d];
  float alpha = 0.0f, beta = 0.0f, eps = -1e30f;
  size_t o = (size_t)b * Tlen * Dd + d;
  for (int t = 0; t < Tlen; ++t, o += Dd) {
    const float kt = Kb[o], vt = Vb[o];
    const float ukt = u + kt;
    const float tau = fmaxf(ukt, eps);
    const float e1 = expf(eps - tau), e2 = expf(ukt - tau);
    const float out = (e1 * alpha + e2 * vt) / (e1 * beta + e2);
    const float weps = eps - w;
    const float tau2 = fmaxf(weps, kt);
    const float d1 = expf(weps - tau2), d2 = expf(kt - tau2);
    alpha = d1 * alpha + d2 * vt;
    beta  = d1 * beta + d2;
    eps   = tau2;
    outs[o] = (__bf16)(out * SR[o]);
  }
}

// ------------------------------ host side ----------------------------------
static void launch_gemm(int mode, const __bf16* A, const __bf16* Bt, void* C,
                        const float* Xin, const float* Aux,
                        int M, int N, int K, hipStream_t s)
{
  dim3 g(N / 64, M / 128), b(256);
  switch (mode) {
    case EP_STORE:   gemm_bf16_wmma<EP_STORE>  <<<g, b, 0, s>>>(A, Bt, C, Xin, Aux, N, K); break;
    case EP_SIGMOID: gemm_bf16_wmma<EP_SIGMOID><<<g, b, 0, s>>>(A, Bt, C, Xin, Aux, N, K); break;
    case EP_RESADD:  gemm_bf16_wmma<EP_RESADD> <<<g, b, 0, s>>>(A, Bt, C, Xin, Aux, N, K); break;
    case EP_RELU2:   gemm_bf16_wmma<EP_RELU2>  <<<g, b, 0, s>>>(A, Bt, C, Xin, Aux, N, K); break;
    default:         gemm_bf16_wmma<EP_FMAOUT> <<<g, b, 0, s>>>(A, Bt, C, Xin, Aux, N, K); break;
  }
}

extern "C" void kernel_launch(void* const* d_in, const int* in_sizes, int n_in,
                              void* d_out, int out_size, void* d_ws, size_t ws_size,
                              hipStream_t stream)
{
  constexpr int Bc = 4, Tc = 4096, Dc = 1024, Fc = 4096;
  constexpr int BT = Bc * Tc;

  const float* x           = (const float*)d_in[0];
  const float* ln1_w       = (const float*)d_in[1];
  const float* ln1_b       = (const float*)d_in[2];
  const float* ln2_w       = (const float*)d_in[3];
  const float* ln2_b       = (const float*)d_in[4];
  const float* time_decay  = (const float*)d_in[5];
  const float* time_first  = (const float*)d_in[6];
  const float* time_mix_k  = (const float*)d_in[7];
  const float* time_mix_v  = (const float*)d_in[8];
  const float* time_mix_r  = (const float*)d_in[9];
  const float* w_key       = (const float*)d_in[10];
  const float* w_value     = (const float*)d_in[11];
  const float* w_recept    = (const float*)d_in[12];
  const float* w_output    = (const float*)d_in[13];
  const float* f_time_mix_k = (const float*)d_in[14];
  const float* f_time_mix_r = (const float*)d_in[15];
  const float* f_w_key     = (const float*)d_in[16];
  const float* f_w_recept  = (const float*)d_in[17];
  const float* f_w_value   = (const float*)d_in[18];
  float* out               = (float*)d_out;

  // ---- workspace carve-up (256B aligned) ----
  char* ws = (char*)d_ws;
  size_t off = 0;
  auto take = [&](size_t bytes) -> char* {
    off = (off + 255) & ~(size_t)255;
    char* p = ws + off;
    off += bytes;
    return p;
  };
  __bf16* wkeyT  = (__bf16*)take((size_t)Dc * Dc * 2);
  __bf16* wvalT  = (__bf16*)take((size_t)Dc * Dc * 2);
  __bf16* wrecT  = (__bf16*)take((size_t)Dc * Dc * 2);
  __bf16* woutT  = (__bf16*)take((size_t)Dc * Dc * 2);
  __bf16* fwrecT = (__bf16*)take((size_t)Dc * Dc * 2);
  __bf16* fwkeyT = (__bf16*)take((size_t)Dc * Fc * 2);   // [F, D]
  __bf16* fwvalT = (__bf16*)take((size_t)Fc * Dc * 2);   // [D, F]
  float*  XN     = (float*) take((size_t)BT * Dc * 4);   // LN output (reused LN2)
  __bf16* MIX    = (__bf16*)take((size_t)BT * Dc * 2);   // mixed operand / wkv*sr
  float*  KB     = (float*) take((size_t)BT * Dc * 4);   // k  (reused as x1)
  float*  VB     = (float*) take((size_t)BT * Dc * 4);   // v
  float*  SRB    = (float*) take((size_t)BT * Dc * 4);   // sigmoid(r) (reused fr)
  __bf16* FK2    = (__bf16*)take((size_t)BT * Fc * 2);   // relu(fk)^2 bf16
  float*  X1     = KB;
  float*  FR     = SRB;
  (void)ws_size; (void)in_sizes; (void)n_in; (void)out_size;

  const size_t totD = (size_t)BT * Dc;

  // ---- 0: weight prep (f32 -> bf16, transposed to [N,K]) ----
  convT_kernel<<<1024, 256, 0, stream>>>(w_key,      wkeyT,  Dc, Dc);
  convT_kernel<<<1024, 256, 0, stream>>>(w_value,    wvalT,  Dc, Dc);
  convT_kernel<<<1024, 256, 0, stream>>>(w_recept,   wrecT,  Dc, Dc);
  convT_kernel<<<1024, 256, 0, stream>>>(w_output,   woutT,  Dc, Dc);
  convT_kernel<<<1024, 256, 0, stream>>>(f_w_recept, fwrecT, Dc, Dc);
  convT_kernel<<<1024, 256, 0, stream>>>(f_w_key,    fwkeyT, Dc, Fc);
  convT_kernel<<<1024, 256, 0, stream>>>(f_w_value,  fwvalT, Fc, Dc);

  // ---- 1: LN1 ----
  ln_kernel<<<BT, 256, 0, stream>>>(x, ln1_w, ln1_b, XN, Dc);

  // ---- 2: k = mix_k @ w_key ----
  mix_kernel<<<4096, 256, 0, stream>>>(XN, time_mix_k, MIX, Tc, Dc, totD);
  launch_gemm(EP_STORE, MIX, wkeyT, KB, nullptr, nullptr, BT, Dc, Dc, stream);
  // ---- 3: v = mix_v @ w_value ----
  mix_kernel<<<4096, 256, 0, stream>>>(XN, time_mix_v, MIX, Tc, Dc, totD);
  launch_gemm(EP_STORE, MIX, wvalT, VB, nullptr, nullptr, BT, Dc, Dc, stream);
  // ---- 4: sr = sigmoid(mix_r @ w_recept) ----
  mix_kernel<<<4096, 256, 0, stream>>>(XN, time_mix_r, MIX, Tc, Dc, totD);
  launch_gemm(EP_SIGMOID, MIX, wrecT, SRB, nullptr, nullptr, BT, Dc, Dc, stream);

  // ---- 5: WKV scan; writes (wkv * sr) as bf16 into MIX ----
  wkv_kernel<<<(Bc * Dc + 255) / 256, 256, 0, stream>>>(
      KB, VB, SRB, time_decay, time_first, MIX, Bc, Tc, Dc);

  // ---- 6: x1 = x + (wkv*sr) @ w_output ----
  launch_gemm(EP_RESADD, MIX, woutT, X1, x, nullptr, BT, Dc, Dc, stream);

  // ---- 7: LN2 ----
  ln_kernel<<<BT, 256, 0, stream>>>(X1, ln2_w, ln2_b, XN, Dc);

  // ---- 8: fr = sigmoid(mix_fr @ f_w_recept) ----
  mix_kernel<<<4096, 256, 0, stream>>>(XN, f_time_mix_r, MIX, Tc, Dc, totD);
  launch_gemm(EP_SIGMOID, MIX, fwrecT, FR, nullptr, nullptr, BT, Dc, Dc, stream);

  // ---- 9: fk2 = relu(mix_fk @ f_w_key)^2 -> bf16 ----
  mix_kernel<<<4096, 256, 0, stream>>>(XN, f_time_mix_k, MIX, Tc, Dc, totD);
  launch_gemm(EP_RELU2, MIX, fwkeyT, FK2, nullptr, nullptr, BT, Fc, Dc, stream);

  // ---- 10: out = x1 + fr * (fk2 @ f_w_value) ----
  launch_gemm(EP_FMAOUT, FK2, fwvalT, out, X1, FR, BT, Dc, Fc, stream);
}
